// GPT_71330816852814
// MI455X (gfx1250) — matrix operand order repaired and checked by reading, not compiled, not run on hardware
//
#include <hip/hip_runtime.h>
#include <cstddef>
#include <cstdint>

// ---------------- model constants ----------------
constexpr int Bc  = 8;
constexpr int Tc  = 1024;
constexpr int Dc  = 1024;
constexpr int Hc  = 16;
constexpr int HSc = 64;      // head size
constexpr int Lc  = 6;
constexpr int Vc  = 16384;
constexpr int Mrows = Bc * Tc;   // 8192

typedef __bf16 bf16_t;
typedef __attribute__((ext_vector_type(16))) __bf16 v16bf;
typedef __attribute__((ext_vector_type(8)))  float  v8f;
typedef __attribute__((ext_vector_type(4)))  unsigned v4u;
typedef __attribute__((ext_vector_type(8)))  int      v8i;
typedef __attribute__((ext_vector_type(4)))  int      v4i;

#if __has_builtin(__builtin_amdgcn_tensor_load_to_lds) && \
    __has_builtin(__builtin_amdgcn_s_wait_tensorcnt)
#define HAVE_TDM 1
#else
#define HAVE_TDM 0
#endif

// ---------------- small helpers ----------------
__device__ __forceinline__ v8f vzero8() {
  v8f z;
#pragma unroll
  for (int i = 0; i < 8; ++i) z[i] = 0.f;
  return z;
}

// Load a 32-byte WMMA operand fragment as two 16-byte chunks.
__device__ __forceinline__ v16bf ld_frag(const bf16_t* p0, const bf16_t* p1) {
  union U { uint4 u[2]; v16bf v; } t;
  t.u[0] = *(const uint4*)p0;
  t.u[1] = *(const uint4*)p1;
  return t.v;
}

__device__ __forceinline__ v8f wmma_bf16(v16bf a, v16bf b, v8f c) {
  return __builtin_amdgcn_wmma_f32_16x16x32_bf16(
      false, a, false, b, (short)0, c, false, false);
}

// reduce across the 16 lanes of a half-wave (xor masks 1,2,4,8 stay in-half)
__device__ __forceinline__ float half_max(float v) {
#pragma unroll
  for (int m = 1; m <= 8; m <<= 1) v = fmaxf(v, __shfl_xor(v, m, 32));
  return v;
}
__device__ __forceinline__ float half_sum(float v) {
#pragma unroll
  for (int m = 1; m <= 8; m <<= 1) v += __shfl_xor(v, m, 32);
  return v;
}

// ---------------- fp32 -> bf16 convert ----------------
__global__ __launch_bounds__(256) void f2bf_kernel(const float* __restrict__ in,
                                                   bf16_t* __restrict__ out,
                                                   long n) {
  long i = ((long)blockIdx.x * 256 + threadIdx.x) * 4;
  if (i + 3 < n) {
    float4 f = *(const float4*)(in + i);
    out[i + 0] = (bf16_t)f.x;
    out[i + 1] = (bf16_t)f.y;
    out[i + 2] = (bf16_t)f.z;
    out[i + 3] = (bf16_t)f.w;
  }
}

// ---------------- embedding: x = tok_emb[idx] + pos_emb ----------------
__global__ __launch_bounds__(256) void embed_kernel(const int* __restrict__ idx,
                                                    const float* __restrict__ tok,
                                                    const float* __restrict__ pos,
                                                    float* __restrict__ x) {
  const int row = blockIdx.x;          // b*T + t
  const int t   = row & (Tc - 1);
  const int tokid = idx[row];
  const float* tr = tok + (size_t)tokid * Dc;
  const float* pr = pos + (size_t)t * Dc;
  float* xr = x + (size_t)row * Dc;
#pragma unroll
  for (int i = 0; i < 4; ++i) {
    int d = threadIdx.x + i * 256;
    xr[d] = tr[d] + pr[d];
  }
}

// ---------------- LayerNorm (f32 in, bf16 out) ----------------
__global__ __launch_bounds__(256) void ln_kernel(const float* __restrict__ x,
                                                 const float* __restrict__ gam,
                                                 const float* __restrict__ bet,
                                                 bf16_t* __restrict__ out) {
  const int row = blockIdx.x, tid = threadIdx.x;
  const float* xr = x + (size_t)row * Dc;
  float vals[4], sum = 0.f, sq = 0.f;
#pragma unroll
  for (int i = 0; i < 4; ++i) {
    float v = xr[tid + i * 256];
    vals[i] = v; sum += v; sq += v * v;
  }
#pragma unroll
  for (int m = 1; m < 32; m <<= 1) {
    sum += __shfl_xor(sum, m, 32);
    sq  += __shfl_xor(sq,  m, 32);
  }
  __shared__ float red[2][8];
  const int lane = tid & 31, wave = tid >> 5;
  if (lane == 0) { red[0][wave] = sum; red[1][wave] = sq; }
  __syncthreads();
  float ts = 0.f, tq = 0.f;
#pragma unroll
  for (int i = 0; i < 8; ++i) { ts += red[0][i]; tq += red[1][i]; }
  const float mean = ts * (1.f / Dc);
  const float var  = tq * (1.f / Dc) - mean * mean;
  const float rs   = rsqrtf(var + 1e-5f);
  bf16_t* orow = out + (size_t)row * Dc;
#pragma unroll
  for (int i = 0; i < 4; ++i) {
    int d = tid + i * 256;
    orow[d] = (bf16_t)((vals[i] - mean) * rs * gam[d] + bet[d]);
  }
}

// ---------------- bf16 WMMA GEMM: C[M,N] = A[M,K] @ W[K,N] (+bias,+resid,relu) ----
// block tile 128x128, 256 threads = 8 waves, wave tile 32(M) x 64(N)
// A tile staged by the Tensor Data Mover (pad_enable reproduces the 40-elem
// LDS row stride: 32 payload bf16 + 8 pad bf16). B tile staged manually
// (transpose, which TDM cannot do).
template <bool OBF, bool RELU>
__global__ __launch_bounds__(256) void gemm_kernel(
    const bf16_t* __restrict__ A, const bf16_t* __restrict__ W,
    const float* __restrict__ bias, const float* __restrict__ resid,
    float* __restrict__ Cf, bf16_t* __restrict__ Cb,
    int M, int N, int K) {
  __shared__ bf16_t As[128 * 40];   // [row][k], stride 40 (80B rows, 16B aligned)
  __shared__ bf16_t Bs[128 * 40];   // transposed: [n][k]
  const int tid  = threadIdx.x;
  const int lane = tid & 31, wave = tid >> 5;
  const int l15  = lane & 15, hi = lane >> 4;
  const int m0 = blockIdx.y * 128;
  const int n0 = blockIdx.x * 128;
  const int waveM = (wave & 3) * 32;   // 4 waves along M
  const int waveN = (wave >> 2) * 64;  // 2 waves along N

  v8f acc[2][4];
#pragma unroll
  for (int i = 0; i < 2; ++i)
#pragma unroll
    for (int j = 0; j < 4; ++j) acc[i][j] = vzero8();

#if HAVE_TDM
  // Constant part of the D# (group 1): data_size=2B, pad_enable,
  // pad_interval=16 DWORDs (code 3), pad_amount=4 DWORDs (code 3),
  // tensor_dim0=K, tensor_dim1=M, tile_dim0=32, tile_dim1=128, stride0=K.
  v8i g1;
  g1[0] = (int)((1u << 16) | (1u << 20) | (3u << 22) | (3u << 25));
  g1[1] = (int)((unsigned)K << 16);
  g1[2] = (int)(((unsigned)K >> 16) | (((unsigned)M & 0xffffu) << 16));
  g1[3] = (int)(((unsigned)M >> 16) | (32u << 16));
  g1[4] = 128;          // tile_dim1
  g1[5] = K;            // tensor_dim0_stride lo
  g1[6] = 0;
  g1[7] = 0;
  v4i gz;
  gz[0] = 0; gz[1] = 0; gz[2] = 0; gz[3] = 0;
  v8i gz8;
#pragma unroll
  for (int i = 0; i < 8; ++i) gz8[i] = 0;
  const unsigned lds_a = (unsigned)(uintptr_t)(void*)As;
#endif

  for (int k0 = 0; k0 < K; k0 += 32) {
#if HAVE_TDM
    if (wave == 0) {
      // D# group 0: count=1, lds_addr, 57-bit global tile address, type=2.
      unsigned long long ga =
          (unsigned long long)(uintptr_t)(A + (size_t)m0 * K + k0);
      v4u g0;
      g0[0] = 1u;
      g0[1] = lds_a;
      g0[2] = (unsigned)(ga & 0xffffffffu);
      g0[3] = (unsigned)((ga >> 32) & 0x01ffffffu) | (2u << 30);
      // 6-arg form on this toolchain (clang-23 / therock-10.0 headers):
      // (g0, g1, g2, g3, <extra group>, cpol). Groups 2/3 + extra = zero (2-D).
      __builtin_amdgcn_tensor_load_to_lds(g0, g1, gz, gz, gz8, 0);
    }
#else
    // fallback: manual A-tile staging (128x32), 16B vector copies
#pragma unroll
    for (int i = 0; i < 2; ++i) {
      int lin = tid + i * 256;            // 0..511
      int row = lin >> 2, seg = (lin & 3) * 8;
      *(uint4*)(As + row * 40 + seg) =
          *(const uint4*)(A + (size_t)(m0 + row) * K + k0 + seg);
    }
#endif
    // prefetch next weight tile while this one is being consumed
    if (k0 + 32 < K)
      __builtin_prefetch(W + (size_t)(k0 + 32) * N + n0 + (tid & 127), 0, 1);
    // stage B tile transposed (32x128 -> [n][k]); overlaps with the TDM DMA
#pragma unroll
    for (int i = 0; i < 4; ++i) {
      int lin = tid + i * 256;            // 0..1023
      int k = lin >> 5, n = (lin & 31) * 4;
      uint2 raw = *(const uint2*)(W + (size_t)(k0 + k) * N + n0 + n);
      bf16_t tmp[4];
      *(uint2*)tmp = raw;
#pragma unroll
      for (int j = 0; j < 4; ++j) Bs[(n + j) * 40 + k] = tmp[j];
    }
#if HAVE_TDM
    if (wave == 0) __builtin_amdgcn_s_wait_tensorcnt(0);
#endif
    __syncthreads();

    v16bf af[2], bfr[4];
#pragma unroll
    for (int mt = 0; mt < 2; ++mt) {
      // A layout: lane row = l15; low lanes K0-15, high lanes K16-31 (contiguous)
      const bf16_t* p = As + (waveM + mt * 16 + l15) * 40 + hi * 16;
      af[mt] = ld_frag(p, p + 8);
    }
#pragma unroll
    for (int nt = 0; nt < 4; ++nt) {
      // B layout: lane col = l15; low lanes K0-15, high lanes K16-31 (contiguous)
      const bf16_t* p = Bs + (waveN + nt * 16 + l15) * 40 + hi * 16;
      bfr[nt] = ld_frag(p, p + 8);
    }
#pragma unroll
    for (int mt = 0; mt < 2; ++mt)
#pragma unroll
      for (int nt = 0; nt < 4; ++nt)
        acc[mt][nt] = wmma_bf16(af[mt], bfr[nt], acc[mt][nt]);
    __syncthreads();
  }

  // epilogue: C layout: VGPR v -> row v + hi*8, lane l15 -> col
#pragma unroll
  for (int mt = 0; mt < 2; ++mt)
#pragma unroll
    for (int nt = 0; nt < 4; ++nt)
#pragma unroll
      for (int v = 0; v < 8; ++v) {
        int r = m0 + waveM + mt * 16 + v + hi * 8;
        int c = n0 + waveN + nt * 16 + l15;
        float val = acc[mt][nt][v];
        if (bias)  val += bias[c];
        if (RELU)  val  = fmaxf(val, 0.f);
        if (resid) val += resid[(size_t)r * N + c];
        if (OBF)   Cb[(size_t)r * N + c] = (bf16_t)val;
        else       Cf[(size_t)r * N + c] = val;
      }
}

// ---------------- flash attention (causal), bf16 in/out ----------------
// grid: (T/64, H, B); 128 threads = 4 waves; wave handles 16 query rows
__global__ __launch_bounds__(128) void attn_kernel(
    const bf16_t* __restrict__ Q, const bf16_t* __restrict__ Km,
    const bf16_t* __restrict__ Vm, bf16_t* __restrict__ Out) {
  __shared__ bf16_t Vs[64 * 72];       // V transposed: [n(hs)][kk(key)]
  __shared__ bf16_t Ps[4][16 * 72];    // per-wave P: [r][c]
  const int tid = threadIdx.x, lane = tid & 31, wave = tid >> 5;
  const int l15 = lane & 15, hi = lane >> 4;
  const int qb = blockIdx.x, h = blockIdx.y, b = blockIdx.z;
  const int q0 = qb * 64 + wave * 16;
  const size_t rowbase = (size_t)b * Tc;
  const int hc = h * HSc;
  const float scale = 0.03125f;        // D^-0.5 quirk (NOT head_size^-0.5)

  // Q fragments (A operand), straight from global; reused for all key tiles
  v16bf qf[2];
#pragma unroll
  for (int kc = 0; kc < 2; ++kc) {
    const bf16_t* p = Q + (rowbase + q0 + l15) * Dc + hc + kc * 32 + hi * 16;
    qf[kc] = ld_frag(p, p + 8);
  }

  v8f o[4];
#pragma unroll
  for (int nt = 0; nt < 4; ++nt) o[nt] = vzero8();
  float mrow[8], lrow[8];
#pragma unroll
  for (int v = 0; v < 8; ++v) { mrow[v] = -3.0e38f; lrow[v] = 0.f; }

  for (int j = 0; j <= qb; ++j) {       // uniform trip count across waves
    const int kt0 = j * 64;
    // cooperative staging of V^T (64 keys x 64 hs)
#pragma unroll
    for (int it = 0; it < 8; ++it) {
      int lin = it * 128 + tid;         // 0..1023, 4 elems each
      int kk = lin >> 4, n4 = (lin & 15) * 4;
      uint2 raw = *(const uint2*)(Vm + (rowbase + kt0 + kk) * Dc + hc + n4);
      bf16_t tmp[4];
      *(uint2*)tmp = raw;
#pragma unroll
      for (int u = 0; u < 4; ++u) Vs[(n4 + u) * 72 + kk] = tmp[u];
    }
    __syncthreads();

    // S = Q @ K^T  (K fragment = B operand, hs-contiguous per lane from global)
    v8f s[4];
#pragma unroll
    for (int nt = 0; nt < 4; ++nt) s[nt] = vzero8();
#pragma unroll
    for (int nt = 0; nt < 4; ++nt)
#pragma unroll
      for (int kc = 0; kc < 2; ++kc) {
        const bf16_t* p =
            Km + (rowbase + kt0 + nt * 16 + l15) * Dc + hc + kc * 32 + hi * 16;
        v16bf kf = ld_frag(p, p + 8);
        s[nt] = wmma_bf16(qf[kc], kf, s[nt]);
      }

    // online softmax, per output row (row stats replicated across half-wave)
#pragma unroll
    for (int v = 0; v < 8; ++v) {
      const int r = q0 + v + hi * 8;
      float sv[4], mx = -3.0e38f;
#pragma unroll
      for (int nt = 0; nt < 4; ++nt) {
        int c = kt0 + nt * 16 + l15;
        float val = s[nt][v] * scale;
        if (c > r) val = -3.0e38f;      // causal mask
        sv[nt] = val;
        mx = fmaxf(mx, val);
      }
      mx = half_max(mx);
      float mnew  = fmaxf(mrow[v], mx);
      float alpha = __expf(mrow[v] - mnew);
      lrow[v] *= alpha;
#pragma unroll
      for (int nt = 0; nt < 4; ++nt) o[nt][v] = o[nt][v] * alpha;
      float ls = 0.f;
#pragma unroll
      for (int nt = 0; nt < 4; ++nt) {
        float p = __expf(sv[nt] - mnew);
        s[nt][v] = p;
        ls += p;
      }
      lrow[v] += half_sum(ls);
      mrow[v] = mnew;
    }

    // transpose P through LDS into A-fragment layout
#pragma unroll
    for (int nt = 0; nt < 4; ++nt)
#pragma unroll
      for (int v = 0; v < 8; ++v)
        Ps[wave][(v + hi * 8) * 72 + nt * 16 + l15] = (bf16_t)s[nt][v];
    __syncthreads();

    v16bf pf[2];
#pragma unroll
    for (int kc = 0; kc < 2; ++kc) {
      const bf16_t* p = &Ps[wave][l15 * 72 + kc * 32 + hi * 16];
      pf[kc] = ld_frag(p, p + 8);
    }
#pragma unroll
    for (int nt = 0; nt < 4; ++nt)
#pragma unroll
      for (int kc = 0; kc < 2; ++kc) {
        const bf16_t* p = Vs + (nt * 16 + l15) * 72 + kc * 32 + hi * 16;
        v16bf vf = ld_frag(p, p + 8);
        o[nt] = wmma_bf16(pf[kc], vf, o[nt]);
      }
    __syncthreads();                    // before Vs is overwritten
  }

  // O / l
#pragma unroll
  for (int v = 0; v < 8; ++v) {
    float inv = 1.f / lrow[v];
#pragma unroll
    for (int nt = 0; nt < 4; ++nt) {
      size_t r = rowbase + q0 + v + hi * 8;
      int c = hc + nt * 16 + l15;
      Out[r * Dc + c] = (bf16_t)(o[nt][v] * inv);
    }
  }
}

// ---------------- host orchestration ----------------
extern "C" void kernel_launch(void* const* d_in, const int* in_sizes, int n_in,
                              void* d_out, int out_size, void* d_ws,
                              size_t ws_size, hipStream_t stream) {
  (void)in_sizes; (void)n_in; (void)out_size; (void)ws_size;

  const int*   idx    = (const int*)  d_in[0];
  const float* tok    = (const float*)d_in[1];
  const float* pos    = (const float*)d_in[2];
  const float* Wq     = (const float*)d_in[3];
  const float* Wk     = (const float*)d_in[4];
  const float* Wv     = (const float*)d_in[5];
  const float* Wo     = (const float*)d_in[6];
  const float* bo     = (const float*)d_in[7];
  const float* ln1_s  = (const float*)d_in[8];
  const float* ln1_b  = (const float*)d_in[9];
  const float* ln2_s  = (const float*)d_in[10];
  const float* ln2_b  = (const float*)d_in[11];
  const float* W1     = (const float*)d_in[12];
  const float* b1     = (const float*)d_in[13];
  const float* W2     = (const float*)d_in[14];
  const float* b2     = (const float*)d_in[15];
  const float* lnf_s  = (const float*)d_in[16];
  const float* lnf_b  = (const float*)d_in[17];
  const float* Wlm    = (const float*)d_in[18];
  const float* blm    = (const float*)d_in[19];
  float* logits = (float*)d_out;

  unsigned char* wsp = (unsigned char*)d_ws;
  size_t off = 0;
  auto alloc = [&](size_t bytes) -> void* {
    void* p = wsp + off;
    off = (off + bytes + 255) & ~(size_t)255;
    return p;
  };
  const size_t DD  = (size_t)Dc * Dc;
  const size_t D4D = (size_t)Dc * 4 * Dc;

  bf16_t* wqb  = (bf16_t*)alloc(Lc * DD * 2);
  bf16_t* wkb  = (bf16_t*)alloc(Lc * DD * 2);
  bf16_t* wvb  = (bf16_t*)alloc(Lc * DD * 2);
  bf16_t* wob  = (bf16_t*)alloc(Lc * DD * 2);
  bf16_t* w1b  = (bf16_t*)alloc(Lc * D4D * 2);
  bf16_t* w2b  = (bf16_t*)alloc(Lc * D4D * 2);
  bf16_t* wlmb = (bf16_t*)alloc((size_t)Dc * Vc * 2);
  float*  x    = (float*) alloc((size_t)Mrows * Dc * 4);
  bf16_t* hbf  = (bf16_t*)alloc((size_t)Mrows * Dc * 2);
  bf16_t* qbf  = (bf16_t*)alloc((size_t)Mrows * Dc * 2);
  bf16_t* kbf  = (bf16_t*)alloc((size_t)Mrows * Dc * 2);
  bf16_t* vbf  = (bf16_t*)alloc((size_t)Mrows * Dc * 2);
  bf16_t* abf  = (bf16_t*)alloc((size_t)Mrows * Dc * 2);
  bf16_t* m1bf = (bf16_t*)alloc((size_t)Mrows * 4 * Dc * 2);

  auto conv = [&](const float* src, bf16_t* dst, size_t n) {
    f2bf_kernel<<<dim3((unsigned)(n / 1024)), 256, 0, stream>>>(src, dst, (long)n);
  };
  conv(Wq,  wqb,  Lc * DD);
  conv(Wk,  wkb,  Lc * DD);
  conv(Wv,  wvb,  Lc * DD);
  conv(Wo,  wob,  Lc * DD);
  conv(W1,  w1b,  Lc * D4D);
  conv(W2,  w2b,  Lc * D4D);
  conv(Wlm, wlmb, (size_t)Dc * Vc);

  embed_kernel<<<Mrows, 256, 0, stream>>>(idx, tok, pos, x);

  const dim3 gD(Dc / 128, Mrows / 128);       // N=1024
  const dim3 g4D(4 * Dc / 128, Mrows / 128);  // N=4096
  const dim3 gV(Vc / 128, Mrows / 128);       // N=16384
  const dim3 gAttn(Tc / 64, Hc, Bc);

  for (int l = 0; l < Lc; ++l) {
    // h = LN1(x) -> bf16
    ln_kernel<<<Mrows, 256, 0, stream>>>(x, ln1_s + (size_t)l * Dc,
                                         ln1_b + (size_t)l * Dc, hbf);
    // q,k,v = h @ W{q,k,v}  (bf16 out)
    gemm_kernel<true, false><<<gD, 256, 0, stream>>>(
        hbf, wqb + l * DD, nullptr, nullptr, nullptr, qbf, Mrows, Dc, Dc);
    gemm_kernel<true, false><<<gD, 256, 0, stream>>>(
        hbf, wkb + l * DD, nullptr, nullptr, nullptr, kbf, Mrows, Dc, Dc);
    gemm_kernel<true, false><<<gD, 256, 0, stream>>>(
        hbf, wvb + l * DD, nullptr, nullptr, nullptr, vbf, Mrows, Dc, Dc);
    // att = causal flash attention
    attn_kernel<<<gAttn, 128, 0, stream>>>(qbf, kbf, vbf, abf);
    // x = x + att @ Wo + bo   (f32, in-place residual)
    gemm_kernel<false, false><<<gD, 256, 0, stream>>>(
        abf, wob + l * DD, bo + (size_t)l * Dc, x, x, nullptr, Mrows, Dc, Dc);
    // h = LN2(x) -> bf16
    ln_kernel<<<Mrows, 256, 0, stream>>>(x, ln2_s + (size_t)l * Dc,
                                         ln2_b + (size_t)l * Dc, hbf);
    // m1 = relu(h @ W1 + b1)  (bf16 out)
    gemm_kernel<true, true><<<g4D, 256, 0, stream>>>(
        hbf, w1b + l * D4D, b1 + (size_t)l * 4 * Dc, nullptr, nullptr, m1bf,
        Mrows, 4 * Dc, Dc);
    // x = x + m1 @ W2 + b2    (f32, in-place residual)
    gemm_kernel<false, false><<<gD, 256, 0, stream>>>(
        m1bf, w2b + l * D4D, b2 + (size_t)l * Dc, x, x, nullptr, Mrows, Dc,
        4 * Dc);
  }

  // final LN + LM head -> logits (f32)
  ln_kernel<<<Mrows, 256, 0, stream>>>(x, lnf_s, lnf_b, hbf);
  gemm_kernel<false, false><<<gV, 256, 0, stream>>>(
      hbf, wlmb, blm, nullptr, logits, nullptr, Mrows, Vc, Dc);
}
